// HQQLinearTorchWeightOnlynt4_80144089743540
// MI455X (gfx1250) — matrix operand order repaired and checked by reading, not compiled, not run on hardware
//
#include <hip/hip_runtime.h>
#include <hip/hip_bf16.h>

typedef __attribute__((ext_vector_type(16))) __bf16 v16bf;
typedef __attribute__((ext_vector_type(8)))  __bf16 v8bf;
typedef __attribute__((ext_vector_type(8)))  float  v8f;

#define KDIM   4096
#define ODIM   11008
#define BDIM   32
#define GRP    128
#define NGRP   32
#define KCHUNK 512
#define NCHUNK (KDIM / KCHUNK)          /* 8 K-segments */
#define LDS_PITCH 520                   /* 512 + 8 bf16 pad: row stride 260 dwords, conflict-free */
#define NPART  (BDIM * ODIM)            /* 352256 f32 partials */

// ---------------------------------------------------------------------------
// Main GEMM. SPLIT=true: one K-segment (512) per block, f32 atomics into pout.
// SPLIT=false: full K per block, direct bf16 epilogue with bias.
// ---------------------------------------------------------------------------
template <bool SPLIT>
__global__ __launch_bounds__(256)
void hqq_int4_wmma_kernel(const __hip_bfloat16* __restrict__ x,
                          const int*            __restrict__ Wq,
                          const __hip_bfloat16* __restrict__ scales,
                          const __hip_bfloat16* __restrict__ zeros,
                          const __hip_bfloat16* __restrict__ bias,
                          float*                __restrict__ pout,
                          __hip_bfloat16*       __restrict__ out)
{
    __shared__ __bf16 ldsx[BDIM * LDS_PITCH];

    const int tid    = threadIdx.x;
    const int lane   = tid & 31;
    const int wave   = tid >> 5;
    const int ncol   = lane & 15;
    const int o      = blockIdx.x * 128 + wave * 16 + ncol;  // output column
    const int aoff   = (lane & 16) ? 8 : 0;    // A: K sub-offset per lane half
    const int koff16 = (lane & 16) ? 8 : 0;    // B: K sub-offset per lane half
    const int arow   = lane & 15;

    const int c0 = SPLIT ? (int)blockIdx.y     : 0;
    const int c1 = SPLIT ? (int)blockIdx.y + 1 : NCHUNK;

    v8f c0acc = {};  // rows 0..15
    v8f c1acc = {};  // rows 16..31

    const long wrowbase = (long)o * KDIM;

    for (int c = c0; c < c1; ++c) {
        __syncthreads();
        // ---- cooperatively stage x[:, c*512 : c*512+512] into LDS (b128 granules)
        {
            const uint4* xg = (const uint4*)x;   // 8 bf16 per uint4
            #pragma unroll
            for (int i = 0; i < 8; ++i) {
                int j    = i * 256 + tid;        // 0..2047
                int row  = j >> 6;               // 64 x 16B per row
                int col8 = j & 63;
                uint4 v = xg[(size_t)row * (KDIM / 8) + (size_t)c * (KCHUNK / 8) + col8];
                *(uint4*)&ldsx[row * LDS_PITCH + col8 * 8] = v;
            }
        }
        __syncthreads();

        #pragma unroll
        for (int gi = 0; gi < KCHUNK / GRP; ++gi) {     // 4 quant groups per chunk
            const int   g  = c * (KCHUNK / GRP) + gi;
            const float s  = __bfloat162float(scales[(size_t)o * NGRP + g]);
            const float zp = __bfloat162float(zeros [(size_t)o * NGRP + g]) - 8.0f * s;

            #pragma unroll
            for (int t = 0; t < GRP / 32; ++t) {        // 4 K-steps of 32
                const int k0   = g  * GRP + t * 32;     // global K offset
                const int kloc = gi * GRP + t * 32;     // chunk-local K offset

                // ---- B fragment: 16 K weights of column o, two runs of 8
                // lanes 0-15 hold K k0+0..7,16..23, lanes 16-31 hold K k0+8..15,24..31
                const int* wp = Wq + wrowbase + k0 + koff16;
                int q[16];
                *(int4*)&q[0]  = *(const int4*)(wp + 0);
                *(int4*)&q[4]  = *(const int4*)(wp + 4);
                *(int4*)&q[8]  = *(const int4*)(wp + 16);
                *(int4*)&q[12] = *(const int4*)(wp + 20);

                v16bf bmat;
                #pragma unroll
                for (int j = 0; j < 16; ++j)
                    bmat[j] = (__bf16)__builtin_fmaf((float)q[j], s, zp);

                // ---- A fragments from LDS (16-bit A layout: 0-7/8-15 | 16-23/24-31)
                const __bf16* a0p = &ldsx[arow * LDS_PITCH + kloc + aoff];
                const __bf16* a1p = &ldsx[(arow + 16) * LDS_PITCH + kloc + aoff];
                v8bf a0lo = *(const v8bf*)(a0p);
                v8bf a0hi = *(const v8bf*)(a0p + 16);
                v8bf a1lo = *(const v8bf*)(a1p);
                v8bf a1hi = *(const v8bf*)(a1p + 16);
                v16bf a0, a1;
                #pragma unroll
                for (int j = 0; j < 8; ++j) {
                    a0[j] = a0lo[j]; a0[j + 8] = a0hi[j];
                    a1[j] = a1lo[j]; a1[j + 8] = a1hi[j];
                }

                // (neg_a, A, neg_b, B, c_mod, C, reuse_a, reuse_b)
                c0acc = __builtin_amdgcn_wmma_f32_16x16x32_bf16(false, a0, false, bmat,
                                                                (short)0, c0acc, false, false);
                c1acc = __builtin_amdgcn_wmma_f32_16x16x32_bf16(false, a1, false, bmat,
                                                                (short)0, c1acc, false, false);
            }
            if (g + 1 < c1 * (KCHUNK / GRP))  // prefetch next group's weights (this block's range)
                __builtin_prefetch(Wq + wrowbase + (size_t)(g + 1) * GRP + koff16, 0, 1);
        }
    }

    const int roff = (lane & 16) ? 8 : 0;
    if (SPLIT) {
        // ---- f32 partial accumulate into scratch (native global_atomic_add_f32)
        #pragma unroll
        for (int v = 0; v < 8; ++v) {
            int b0 = v + roff;
            __hip_atomic_fetch_add(&pout[(size_t)b0 * ODIM + o], c0acc[v],
                                   __ATOMIC_RELAXED, __HIP_MEMORY_SCOPE_AGENT);
            __hip_atomic_fetch_add(&pout[(size_t)(b0 + 16) * ODIM + o], c1acc[v],
                                   __ATOMIC_RELAXED, __HIP_MEMORY_SCOPE_AGENT);
        }
    } else {
        // ---- direct epilogue: bias add, f32 -> bf16
        const float bv = __bfloat162float(bias[o]);
        #pragma unroll
        for (int v = 0; v < 8; ++v) {
            int b0 = v + roff;
            out[(size_t)b0 * ODIM + o]        = __float2bfloat16(c0acc[v] + bv);
            out[(size_t)(b0 + 16) * ODIM + o] = __float2bfloat16(c1acc[v] + bv);
        }
    }
}

// ---------------------------------------------------------------------------
// Zero the f32 partial buffer (uint4 per thread; 344*256*4 = 352256 floats)
// ---------------------------------------------------------------------------
__global__ __launch_bounds__(256)
void hqq_zero_kernel(float* __restrict__ pout)
{
    size_t i = (size_t)blockIdx.x * 256 + threadIdx.x;   // 88064 threads
    uint4 z = {0u, 0u, 0u, 0u};
    ((uint4*)pout)[i] = z;
}

// ---------------------------------------------------------------------------
// Finalize: out[b][o] = bf16(partial[b][o] + bias[o])
// ---------------------------------------------------------------------------
__global__ __launch_bounds__(256)
void hqq_finalize_kernel(const float* __restrict__ pout,
                         const __hip_bfloat16* __restrict__ bias,
                         __hip_bfloat16* __restrict__ out)
{
    size_t i = (size_t)blockIdx.x * 256 + threadIdx.x;   // 88064 threads, 4 elems each
    float4 p = ((const float4*)pout)[i];
    int obase = (int)((i * 4) % ODIM);                   // ODIM % 4 == 0 -> same row
    __hip_bfloat16 r[4];
    r[0] = __float2bfloat16(p.x + __bfloat162float(bias[obase + 0]));
    r[1] = __float2bfloat16(p.y + __bfloat162float(bias[obase + 1]));
    r[2] = __float2bfloat16(p.z + __bfloat162float(bias[obase + 2]));
    r[3] = __float2bfloat16(p.w + __bfloat162float(bias[obase + 3]));
    *(uint2*)&out[i * 4] = *(uint2*)r;
}

extern "C" void kernel_launch(void* const* d_in, const int* in_sizes, int n_in,
                              void* d_out, int out_size, void* d_ws, size_t ws_size,
                              hipStream_t stream) {
    const __hip_bfloat16* x      = (const __hip_bfloat16*)d_in[0];
    const int*            Wq     = (const int*)d_in[1];
    const __hip_bfloat16* scales = (const __hip_bfloat16*)d_in[2];
    const __hip_bfloat16* zeros  = (const __hip_bfloat16*)d_in[3];
    const __hip_bfloat16* bias   = (const __hip_bfloat16*)d_in[4];
    __hip_bfloat16*       out    = (__hip_bfloat16*)d_out;

    const bool split = ws_size >= (size_t)NPART * sizeof(float);

    if (split) {
        float* pws = (float*)d_ws;
        // 1) zero partials
        hipLaunchKernelGGL(hqq_zero_kernel, dim3(NPART / 1024), dim3(256), 0, stream, pws);
        // 2) K-split GEMM: 86 column strips x 8 K-segments = 688 workgroups
        hipLaunchKernelGGL((hqq_int4_wmma_kernel<true>), dim3(ODIM / 128, NCHUNK), dim3(256),
                           0, stream, x, Wq, scales, zeros, bias, pws, out);
        // 3) bias + convert
        hipLaunchKernelGGL(hqq_finalize_kernel, dim3(NPART / 1024), dim3(256), 0, stream,
                           pws, bias, out);
    } else {
        // fallback: monolithic kernel, one 8-wave workgroup per 128-column strip over full K
        hipLaunchKernelGGL((hqq_int4_wmma_kernel<false>), dim3(ODIM / 128, 1), dim3(256),
                           0, stream, x, Wq, scales, zeros, bias, (float*)nullptr, out);
    }
}
